// GraphMatch_65867618451820
// MI455X (gfx1250) — compile-verified, matrix operand order checked by reading, not run on hardware
//
#include <hip/hip_runtime.h>
#include <math.h>

#define N_NODES 256
#define DFEAT   128
#define HID     512

typedef __attribute__((ext_vector_type(16))) _Float16 v16h;
typedef __attribute__((ext_vector_type(8)))  float    v8f;

union V16H { v16h v; unsigned int u[8]; };

// ---------------------------------------------------------------------------
// WMMA A-fragment loader: 16x32 f16 tile, row-major in LDS, rowStride halves.
// Lane layout (wave32): lane = g*16+m holds row m; halves h: K = (h&7)+(h>>3)*16+g*8.
// Consecutive half-pairs are consecutive K -> 8x ds_read_b32 per fragment.
// rowStride is padded (520 / 136) so the 16 lanes of a fetch hit distinct banks.
// ---------------------------------------------------------------------------
__device__ inline v16h load_afrag(const _Float16* tile, int rowStride, int k0, int lane) {
    int g = lane >> 4, m = lane & 15;
    const _Float16* row = tile + m * rowStride + k0 + (g << 3);
    V16H a;
#pragma unroll
    for (int t = 0; t < 8; ++t) {
        int K = (t < 4) ? (2 * t) : (16 + 2 * (t - 4));
        a.u[t] = *(const unsigned int*)(row + K);
    }
    return a.v;
}

__device__ inline v8f wmma_step(v16h a, v16h b, v8f c) {
    return __builtin_amdgcn_wmma_f32_16x16x32_f16(false, a, false, b, (short)0, c, false, false);
}

// ---------------------------------------------------------------------------
// Pack ee_w2 [512,128] f32 into B-fragment order (8 colgroups x 16 ksteps x 32 lanes x 16 halves)
// ---------------------------------------------------------------------------
__global__ void pack_w2frag(const float* __restrict__ w2, _Float16* __restrict__ dst) {
    int idx = blockIdx.x * 256 + threadIdx.x;          // 65536
    int h = idx & 15, lane = (idx >> 4) & 31, ks = (idx >> 9) & 15, cg = idx >> 13;
    int g = lane >> 4, n = lane & 15;
    int K = (ks << 5) + (h & 7) + ((h >> 3) << 4) + (g << 3);
    dst[idx] = (_Float16)w2[K * 128 + (cg << 4) + n];
}

// Pack We = pr_w1 rows 256..383 ([128,512]) into 32 colgroups x 4 ksteps x 32 x 16
__global__ void pack_wpfrag(const float* __restrict__ w1, _Float16* __restrict__ dst) {
    int idx = blockIdx.x * 256 + threadIdx.x;          // 65536
    int h = idx & 15, lane = (idx >> 4) & 31, ks = (idx >> 9) & 3, cgg = idx >> 11;
    int g = lane >> 4, n = lane & 15;
    int K = (ks << 5) + (h & 7) + ((h >> 3) << 4) + (g << 3);
    dst[idx] = (_Float16)w1[(256 + K) * 512 + (cgg << 4) + n];
}

// ---------------------------------------------------------------------------
// Generic small GEMM: C[MxN] = alpha*(A[MxK] @ B[KxN]) + bias + Add, opt relu
// ---------------------------------------------------------------------------
__global__ void gemm_naive(const float* __restrict__ A, const float* __restrict__ B,
                           const float* __restrict__ bias, const float* __restrict__ Add,
                           float* __restrict__ C, int M, int K, int N, float alpha, int relu) {
    int idx = blockIdx.x * blockDim.x + threadIdx.x;
    if (idx >= M * N) return;
    int m = idx / N, nc = idx - m * N;
    const float* a = A + m * K;
    const float* b = B + nc;
    float s = 0.f;
    for (int k = 0; k < K; ++k) s += a[k] * b[k * N];
    s *= alpha;
    if (bias) s += bias[nc];
    if (Add)  s += Add[idx];
    if (relu) s = s > 0.f ? s : 0.f;
    C[idx] = s;
}

// ---------------------------------------------------------------------------
// Edge embedding layer 2 (the big one):
//   E[i,j,:] = (relu(A[i]+B[j]) [512]) @ ee_w2[512,128] + ee_b2   (A has b1 folded)
// block = (i0, 16 j's); 8 waves, wave w -> out cols [16w,16w+16).
// Whole h1 tile staged once (1 barrier), 16 K-steps back-to-back, 2 accumulators.
// ---------------------------------------------------------------------------
__global__ void edge2_wmma(const float* __restrict__ Ae, const float* __restrict__ Be,
                           const _Float16* __restrict__ w2f, const float* __restrict__ eb2,
                           _Float16* __restrict__ E) {
    __shared__ __align__(32) _Float16 h1[16 * 520];    // padded row stride vs bank conflicts
    int i0 = blockIdx.y;
    int j0 = blockIdx.x << 4;
    int tid = threadIdx.x;
    int w = tid >> 5, lane = tid & 31;
    int g = lane >> 4, n = lane & 15;
    int c = (w << 4) + n;
    float bias2 = eb2[c];
    {   // build relu(A[i0]+B[j]) f16 tile: 16 threads per row, 32 contiguous K each
        int r = tid >> 4;
        int col0 = (tid & 15) << 5;
        const float* arow = Ae + i0 * 512 + col0;
        const float* brow = Be + (j0 + r) * 512 + col0;
        _Float16* dst = h1 + r * 520 + col0;
#pragma unroll
        for (int k = 0; k < 32; k += 2) {
            float v0 = arow[k] + brow[k];
            float v1 = arow[k + 1] + brow[k + 1];
            v0 = v0 > 0.f ? v0 : 0.f;
            v1 = v1 > 0.f ? v1 : 0.f;
            dst[k]     = (_Float16)v0;
            dst[k + 1] = (_Float16)v1;
        }
    }
    __syncthreads();
    v8f acc0 = {}, acc1 = {};
#pragma unroll
    for (int ks = 0; ks < 16; ks += 2) {
        v16h a0 = load_afrag(h1, 520, ks << 5, lane);
        v16h b0 = *(const v16h*)(w2f + (((w << 4) + ks) * 32 + lane) * 16);
        acc0 = wmma_step(a0, b0, acc0);
        v16h a1 = load_afrag(h1, 520, (ks + 1) << 5, lane);
        v16h b1 = *(const v16h*)(w2f + (((w << 4) + ks + 1) * 32 + lane) * 16);
        acc1 = wmma_step(a1, b1, acc1);
    }
#pragma unroll
    for (int r = 0; r < 8; ++r) {
        int j = j0 + r + (g << 3);
        E[(((i0 << 8) + j) << 7) + c] = (_Float16)(acc0[r] + acc1[r] + bias2);
    }
}

// ---------------------------------------------------------------------------
// Propagation fused GEMM + reduction:
//   S[m,c] = sum_n relu( E[m,n,:]@We[:,c] + C[m,c] + R[n,c] )
// block = (colgroup of 128, m-tile of 16); 8 waves -> 16 cols each.
// 4 n-tiles staged per barrier (16 WMMAs between barriers); prefetch next chunk.
// ---------------------------------------------------------------------------
#define ET_STRIDE 136
#define ET_TILE   (16 * ET_STRIDE)
__global__ void prop_wmma(const _Float16* __restrict__ E, const _Float16* __restrict__ wpf,
                          const float* __restrict__ Cp, const float* __restrict__ Rp,
                          float* __restrict__ S) {
    __shared__ __align__(32) _Float16 wp[8 * 4 * 32 * 16];   // 32 KB: this block's We frags
    __shared__ __align__(32) _Float16 et[4 * ET_TILE];       // 17 KB: 4 E tiles (padded)
    int cg4 = blockIdx.x;
    int m0 = blockIdx.y << 4;
    int tid = threadIdx.x;
    int w = tid >> 5, lane = tid & 31, g = lane >> 4, n = lane & 15;
    int c = (cg4 << 7) + (w << 4) + n;
    {   // stage We fragments (contiguous 32KB chunk) into LDS
        const int4* src = (const int4*)(wpf + (cg4 << 14));
        int4* dst = (int4*)wp;
        for (int t = tid; t < 2048; t += 256) dst[t] = src[t];
    }
    float creg[8];
#pragma unroll
    for (int r = 0; r < 8; ++r) creg[r] = Cp[(m0 + r + (g << 3)) * 512 + c];
    float sacc[8] = {};
    __syncthreads();
    int r_ld = tid >> 4, seg = tid & 15;                     // cooperative-load coords
    for (int nn0 = 0; nn0 < 256; nn0 += 4) {
#pragma unroll
        for (int it = 0; it < 4; ++it) {                     // 4 E tiles -> LDS
            const char* src = (const char*)(E + ((((m0 + r_ld) << 8) + nn0 + it) << 7) + (seg << 3));
            *(int4*)(et + it * ET_TILE + r_ld * ET_STRIDE + (seg << 3)) = *(const int4*)src;
            if (nn0 < 252) __builtin_prefetch(src + 1024, 0, 1);   // next chunk (+4 n's)
        }
        __syncthreads();
#pragma unroll
        for (int it = 0; it < 4; ++it) {
            int nn = nn0 + it;
            float rv = Rp[(nn << 9) + c];
            const _Float16* tp = et + it * ET_TILE;
            v8f accA = {}, accB = {};
            accA = wmma_step(load_afrag(tp, ET_STRIDE, 0,  lane),
                             *(const v16h*)(wp + (((w << 2) + 0) * 32 + lane) * 16), accA);
            accB = wmma_step(load_afrag(tp, ET_STRIDE, 32, lane),
                             *(const v16h*)(wp + (((w << 2) + 1) * 32 + lane) * 16), accB);
            accA = wmma_step(load_afrag(tp, ET_STRIDE, 64, lane),
                             *(const v16h*)(wp + (((w << 2) + 2) * 32 + lane) * 16), accA);
            accB = wmma_step(load_afrag(tp, ET_STRIDE, 96, lane),
                             *(const v16h*)(wp + (((w << 2) + 3) * 32 + lane) * 16), accB);
#pragma unroll
            for (int r = 0; r < 8; ++r) {
                float v = accA[r] + accB[r] + creg[r] + rv;
                sacc[r] += (v > 0.f ? v : 0.f);
            }
        }
        __syncthreads();
    }
#pragma unroll
    for (int r = 0; r < 8; ++r) S[(m0 + r + (g << 3)) * 512 + c] = sacc[r];
}

// ---------------------------------------------------------------------------
// Cross-graph interaction: sum_j p[i,j,c] = a[i,c]*Sb[c]  ->  per-(i,c) loop over j.
// ---------------------------------------------------------------------------
__global__ void interact_k(const float* __restrict__ A, const float* __restrict__ B,
                           float* __restrict__ out) {
    int i = blockIdx.x, c = threadIdx.x;
    float a = A[i * 128 + c];
    float sb = 0.f, num = 0.f;
    for (int j = 0; j < 256; ++j) {
        float bv = B[j * 128 + c];
        sb += bv;
        num += bv * __expf(a * bv);
    }
    out[i * 128 + c] = num / (a * sb) * (1.0f / 256.f);
}

// ---------------------------------------------------------------------------
// Z[i] = [l2norm(emb[i]) | l2norm(intra[i]) | l2norm(cross[i])]  (256 x 384)
// ---------------------------------------------------------------------------
__global__ void build_z(const float* __restrict__ emb, const float* __restrict__ intra,
                        const float* __restrict__ cross, float* __restrict__ Z) {
    __shared__ float red[3][4];
    int i = blockIdx.x, c = threadIdx.x;     // 128 threads
    int w = c >> 5, lane = c & 31;
    float e = emb[i * 128 + c], t = intra[i * 128 + c], x = cross[i * 128 + c];
    float se = e * e, st = t * t, sx = x * x;
#pragma unroll
    for (int o = 16; o; o >>= 1) {
        se += __shfl_down(se, o);
        st += __shfl_down(st, o);
        sx += __shfl_down(sx, o);
    }
    if (lane == 0) { red[0][w] = se; red[1][w] = st; red[2][w] = sx; }
    __syncthreads();
    float ne = sqrtf(red[0][0] + red[0][1] + red[0][2] + red[0][3]);
    float nt = sqrtf(red[1][0] + red[1][1] + red[1][2] + red[1][3]);
    float nx = sqrtf(red[2][0] + red[2][1] + red[2][2] + red[2][3]);
    Z[i * 384 + c]       = e / fmaxf(ne, 1e-12f);
    Z[i * 384 + 128 + c] = t / fmaxf(nt, 1e-12f);
    Z[i * 384 + 256 + c] = x / fmaxf(nx, 1e-12f);
}

// ---------------------------------------------------------------------------
// Aggregation: s = softmax_i(upd@agw[:128] + mean(upd)@agw[128:] + agb); r = sum_i s_i*upd[i]
// ---------------------------------------------------------------------------
__global__ void aggregate_k(const float* __restrict__ upd, const float* __restrict__ agw,
                            const float* __restrict__ agb, float* __restrict__ r) {
    __shared__ float sv[256];
    __shared__ float red[256];
    __shared__ float mu[128];
    int t = threadIdx.x;                      // 256 threads
    if (t < 128) {
        float s = 0.f;
        for (int i = 0; i < 256; ++i) s += upd[i * 128 + t];
        mu[t] = s * (1.f / 256.f);
    }
    __syncthreads();
    red[t] = (t < 128) ? mu[t] * agw[128 + t] : 0.f;
    __syncthreads();
    for (int o = 128; o > 0; o >>= 1) { if (t < o) red[t] += red[t + o]; __syncthreads(); }
    float Kc = red[0] + agb[0];
    __syncthreads();
    float l = 0.f;
    for (int c = 0; c < 128; ++c) l += upd[t * 128 + c] * agw[c];
    l += Kc;
    sv[t] = l; red[t] = l;
    __syncthreads();
    for (int o = 128; o > 0; o >>= 1) { if (t < o) red[t] = fmaxf(red[t], red[t + o]); __syncthreads(); }
    float mx = red[0];
    __syncthreads();
    float e = __expf(sv[t] - mx);
    sv[t] = e; red[t] = e;
    __syncthreads();
    for (int o = 128; o > 0; o >>= 1) { if (t < o) red[t] += red[t + o]; __syncthreads(); }
    float inv = 1.f / red[0];
    __syncthreads();
    sv[t] *= inv;
    __syncthreads();
    if (t < 128) {
        float acc = 0.f;
        for (int i = 0; i < 256; ++i) acc += sv[i] * upd[i * 128 + t];
        r[t] = acc;
    }
}

// Final cosine similarity -> (cos+1)/2
__global__ void final_k(const float* __restrict__ r1, const float* __restrict__ r2,
                        float* __restrict__ out) {
    __shared__ float red[3][4];
    int c = threadIdx.x, w = c >> 5, lane = c & 31;  // 128 threads
    float a = r1[c], b = r2[c];
    float sn = a * b, sa = a * a, sb = b * b;
#pragma unroll
    for (int o = 16; o; o >>= 1) {
        sn += __shfl_down(sn, o);
        sa += __shfl_down(sa, o);
        sb += __shfl_down(sb, o);
    }
    if (lane == 0) { red[0][w] = sn; red[1][w] = sa; red[2][w] = sb; }
    __syncthreads();
    if (c == 0) {
        float num = red[0][0] + red[0][1] + red[0][2] + red[0][3];
        float na = sqrtf(red[1][0] + red[1][1] + red[1][2] + red[1][3]);
        float nb = sqrtf(red[2][0] + red[2][1] + red[2][2] + red[2][3]);
        float den = fmaxf(na * nb, 1e-8f);
        out[0] = (num / den + 1.f) * 0.5f;
    }
}

// ---------------------------------------------------------------------------
extern "C" void kernel_launch(void* const* d_in, const int* in_sizes, int n_in,
                              void* d_out, int out_size, void* d_ws, size_t ws_size,
                              hipStream_t stream) {
    (void)in_sizes; (void)n_in; (void)out_size; (void)ws_size;
    const float* x[2]   = { (const float*)d_in[0], (const float*)d_in[1] };
    const float* ne_w1  = (const float*)d_in[2];
    const float* ne_b1  = (const float*)d_in[3];
    const float* ne_w2  = (const float*)d_in[4];
    const float* ne_b2  = (const float*)d_in[5];
    const float* ee_w1  = (const float*)d_in[6];
    const float* ee_b1  = (const float*)d_in[7];
    const float* ee_w2  = (const float*)d_in[8];
    const float* ee_b2  = (const float*)d_in[9];
    const float* pr_w1  = (const float*)d_in[10];
    const float* pr_b1  = (const float*)d_in[11];
    const float* pr_w2  = (const float*)d_in[12];
    const float* pr_b2  = (const float*)d_in[13];
    const float* up_w   = (const float*)d_in[14];
    const float* up_b   = (const float*)d_in[15];
    const float* ag_w   = (const float*)d_in[16];
    const float* ag_b   = (const float*)d_in[17];

    char* base = (char*)d_ws;
    size_t off = 0;
    auto alloc = [&](size_t bytes) -> char* {
        off = (off + 255) & ~(size_t)255;
        char* p = base + off;
        off += bytes;
        return p;
    };

    float* t1 = (float*)alloc(256 * 512 * 4);
    float *emb[2], *Ae[2], *Be[2], *Cp[2], *Rp[2], *Sb[2], *intra[2], *cross[2], *Zb[2], *updb[2], *rv[2];
    _Float16 *Eb[2];
    for (int g = 0; g < 2; ++g) {
        emb[g]   = (float*)alloc(256 * 128 * 4);
        Ae[g]    = (float*)alloc(256 * 512 * 4);
        Be[g]    = (float*)alloc(256 * 512 * 4);
        Cp[g]    = (float*)alloc(256 * 512 * 4);
        Rp[g]    = (float*)alloc(256 * 512 * 4);
        Sb[g]    = (float*)alloc(256 * 512 * 4);
        intra[g] = (float*)alloc(256 * 128 * 4);
        cross[g] = (float*)alloc(256 * 128 * 4);
        Zb[g]    = (float*)alloc(256 * 384 * 4);
        updb[g]  = (float*)alloc(256 * 128 * 4);
        rv[g]    = (float*)alloc(128 * 4);
    }
    _Float16* w2f = (_Float16*)alloc(65536 * 2);
    _Float16* wpf = (_Float16*)alloc(65536 * 2);
    for (int g = 0; g < 2; ++g) Eb[g] = (_Float16*)alloc((size_t)256 * 256 * 128 * 2);

    auto gemm = [&](const float* A, const float* B, const float* bias, const float* Add,
                    float* C, int M, int K, int N, float alpha, int relu) {
        int total = M * N;
        gemm_naive<<<(total + 255) / 256, 256, 0, stream>>>(A, B, bias, Add, C, M, K, N, alpha, relu);
    };

    pack_w2frag<<<256, 256, 0, stream>>>(ee_w2, w2f);
    pack_wpfrag<<<256, 256, 0, stream>>>(pr_w1, wpf);

    for (int g = 0; g < 2; ++g) {
        // node embedding MLP
        gemm(x[g], ne_w1, ne_b1, nullptr, t1, 256, 128, 512, 1.f, 1);
        gemm(t1, ne_w2, ne_b2, nullptr, emb[g], 256, 512, 128, 1.f, 0);
        // decomposed edge layer-1 halves: A = x@Wx1 + W_oh_i + b1 ; B = x@Wx2 + W_oh_j
        gemm(x[g], ee_w1,             ee_b1,   ee_w1 + 128 * 512, Ae[g], 256, 128, 512, 1.f, 0);
        gemm(x[g], ee_w1 + 384 * 512, nullptr, ee_w1 + 512 * 512, Be[g], 256, 128, 512, 1.f, 0);
        // decomposed propagation layer-1 row/col terms
        gemm(x[g], pr_w1,             pr_b1,   nullptr, Cp[g], 256, 128, 512, 1.f, 0);
        gemm(x[g], pr_w1 + 128 * 512, nullptr, nullptr, Rp[g], 256, 128, 512, 1.f, 0);
        // heavy WMMA stages
        edge2_wmma<<<dim3(16, 256), 256, 0, stream>>>(Ae[g], Be[g], w2f, ee_b2, Eb[g]);
        prop_wmma<<<dim3(4, 16), 256, 0, stream>>>(Eb[g], wpf, Cp[g], Rp[g], Sb[g]);
        // intra = (S/256) @ pr_w2 + pr_b2  (mean folded through linear layer)
        gemm(Sb[g], pr_w2, pr_b2, nullptr, intra[g], 256, 512, 128, 1.f / 256.f, 0);
    }

    interact_k<<<256, 128, 0, stream>>>(emb[0], emb[1], cross[0]);
    interact_k<<<256, 128, 0, stream>>>(emb[1], emb[0], cross[1]);

    for (int g = 0; g < 2; ++g) {
        build_z<<<256, 128, 0, stream>>>(emb[g], intra[g], cross[g], Zb[g]);
        gemm(Zb[g], up_w, up_b, nullptr, updb[g], 256, 384, 128, 1.f, 0);
        aggregate_k<<<1, 256, 0, stream>>>(updb[g], ag_w, ag_b, rv[g]);
    }
    final_k<<<1, 128, 0, stream>>>(rv[0], rv[1], (float*)d_out);
}
